// CharDecoder_61890478735521
// MI455X (gfx1250) — matrix operand order repaired and checked by reading, not compiled, not run on hardware
//
#include <hip/hip_runtime.h>
#include <stdint.h>

#define T_STEPS 21
#define BATCH   4096
#define HID     1024
#define EMB     50
#define EPAD    64
#define VOC     96
#define G4      4096   // 4*HID

typedef __attribute__((ext_vector_type(16))) __bf16 v16bf;
typedef __attribute__((ext_vector_type(8)))  __bf16 v8bf;
typedef __attribute__((ext_vector_type(8)))  float  v8f;
typedef int v2i __attribute__((vector_size(8)));

// global-address-space views to force global_load (not flat_load) codegen
typedef __attribute__((address_space(1))) const v16bf gv16bf;
typedef __attribute__((address_space(1))) const v8bf  gv8bf;

#if defined(__gfx1250__) && __has_builtin(__builtin_amdgcn_global_load_async_to_lds_b64)
#define HAVE_ASYNC_LDS 1
typedef __attribute__((address_space(1))) v2i gv2i;   // global b64 payload
typedef __attribute__((address_space(3))) v2i lv2i;   // LDS b64 payload
#endif

__device__ __forceinline__ float sigmoidf_(float x) {
    return 1.0f / (1.0f + __expf(-x));
}
__device__ __forceinline__ float tanhf_(float x) {
    float xc = fminf(fmaxf(x, -15.0f), 15.0f);
    float e  = __expf(2.0f * xc);
    return (e - 1.0f) / (e + 1.0f);
}
__device__ __forceinline__ v16bf cat8(v8bf lo, v8bf hi) {
    return __builtin_shufflevector(lo, hi, 0,1,2,3,4,5,6,7,8,9,10,11,12,13,14,15);
}

__device__ __forceinline__ void cp_b64_to_lds(__bf16* ldst, const __bf16* gsrc) {
#ifdef HAVE_ASYNC_LDS
    __builtin_amdgcn_global_load_async_to_lds_b64((gv2i*)gsrc, (lv2i*)ldst, 0, 0);
#else
    *(uint64_t*)ldst = *(const uint64_t*)gsrc;
#endif
}
__device__ __forceinline__ void wait_async_lds() {
#ifdef HAVE_ASYNC_LDS
 #if __has_builtin(__builtin_amdgcn_s_wait_asynccnt)
    __builtin_amdgcn_s_wait_asynccnt(0);
 #else
    asm volatile("s_wait_asynccnt 0x0" ::: "memory");
 #endif
#endif
}

// ---------------- setup kernels ----------------
__global__ void k_cvt_bf16(const float* __restrict__ s, uint16_t* __restrict__ d, int n) {
    int i = blockIdx.x * blockDim.x + threadIdx.x;
    if (i < n) ((__bf16*)d)[i] = (__bf16)s[i];
}
__global__ void k_pad_wih(const float* __restrict__ s, uint16_t* __restrict__ d) {
    int i = blockIdx.x * blockDim.x + threadIdx.x;   // G4*EPAD
    if (i < G4 * EPAD) {
        int g = i / EPAD, e = i % EPAD;
        ((__bf16*)d)[i] = (e < EMB) ? (__bf16)s[g * EMB + e] : (__bf16)0.0f;
    }
}
__global__ void k_bias(const float* __restrict__ a, const float* __restrict__ b,
                       float* __restrict__ d) {
    int i = blockIdx.x * blockDim.x + threadIdx.x;
    if (i < G4) d[i] = a[i] + b[i];
}
__global__ void k_embed(const int* __restrict__ ids, const float* __restrict__ emb,
                        uint16_t* __restrict__ d) {
    int i = blockIdx.x * blockDim.x + threadIdx.x;   // T*B*EPAD
    if (i < T_STEPS * BATCH * EPAD) {
        int e = i % EPAD, tb = i / EPAD;
        int id = ids[tb];
        ((__bf16*)d)[i] = (e < EMB) ? (__bf16)emb[id * EMB + e] : (__bf16)0.0f;
    }
}
__global__ void k_init(const float* __restrict__ h0, const float* __restrict__ c0,
                       uint16_t* __restrict__ hp, float* __restrict__ cf) {
    int i = blockIdx.x * blockDim.x + threadIdx.x;
    if (i < BATCH * HID) { ((__bf16*)hp)[i] = (__bf16)h0[i]; cf[i] = c0[i]; }
}
__global__ void k_final(const float* __restrict__ hf, const float* __restrict__ cf,
                        float* __restrict__ out) {
    int i = blockIdx.x * blockDim.x + threadIdx.x;
    if (i < BATCH * HID) { out[i] = hf[i]; out[BATCH * HID + i] = cf[i]; }
}

// one K=32 chunk: A frags from LDS tile, B frags from global (AS1), 8 WMMAs
__device__ __forceinline__ void mma_chunk(const __bf16* tile,
                                          const __bf16* const (&bp)[4],
                                          v8f (&acc)[2][4], int l16, int half) {
    v16bf afrag[2];
    #pragma unroll
    for (int mf = 0; mf < 2; ++mf) {
        const __bf16* ab = tile + (mf * 16 + l16) * 56;
        v8bf lo = *(const v8bf*)(ab + 16 * half);
        v8bf hi = *(const v8bf*)(ab + 16 * half + 8);
        afrag[mf] = cat8(lo, hi);
    }
    #pragma unroll
    for (int g = 0; g < 4; ++g) {
        v16bf bfrag = *(gv16bf*)bp[g];
        #pragma unroll
        for (int mf = 0; mf < 2; ++mf)
            acc[mf][g] = __builtin_amdgcn_wmma_f32_16x16x32_bf16(
                false, afrag[mf], false, bfrag, (short)0, acc[mf][g], false, false);
    }
}

// ---------------- fused LSTM step: gates GEMM (WMMA bf16) + cell ----------------
// grid = (BATCH/32, HID/128), block = 256 (8 waves). Wave w: 16 h-cols, M=32 rows.
// Double-buffered LDS A-tile, async copy-in overlapped with WMMA of current tile.
__global__ __launch_bounds__(256) void k_lstm_step(
    const uint16_t* __restrict__ Whh_u, const uint16_t* __restrict__ Wih_u,
    const float*    __restrict__ bias,
    const uint16_t* __restrict__ xt_u,  const uint16_t* __restrict__ hin_u,
    uint16_t* __restrict__ hout_u, float* __restrict__ hf32, float* __restrict__ cf32)
{
    const __bf16* Whh = (const __bf16*)Whh_u;
    const __bf16* Wih = (const __bf16*)Wih_u;
    const __bf16* xt  = (const __bf16*)xt_u;
    const __bf16* hin = (const __bf16*)hin_u;
    __bf16* hout = (__bf16*)hout_u;

    // 2 buffers x 32 rows, row stride 56 bf16 (112B: 16B-aligned, conflict-free b128)
    __shared__ __align__(16) __bf16 tileA[2][32 * 56];

    const int tid  = threadIdx.x;
    const int wave = tid >> 5;
    const int lane = tid & 31;
    const int l16  = lane & 15;
    const int half = lane >> 4;
    const int mBase = blockIdx.x * 32;
    const int nTile = blockIdx.y * 128 + wave * 16;
    const int ncol  = nTile + l16;                    // h column for this lane

    const int ldRow = tid >> 3;                       // 0..31
    const int ldQ   = (tid & 7) * 4;                  // 4 bf16 per thread
    const int ldOff = ldRow * 56 + ldQ;               // LDS dest offset (elements)

    // per-thread source pointers for the cooperative A-tile copy
    const __bf16* srcH = hin + (size_t)(mBase + ldRow) * HID  + ldQ;
    const __bf16* srcX = xt  + (size_t)(mBase + ldRow) * EPAD + ldQ;

    v8f acc[2][4];
    const v8f zero = {0.f,0.f,0.f,0.f,0.f,0.f,0.f,0.f};
    #pragma unroll
    for (int mf = 0; mf < 2; ++mf)
        #pragma unroll
        for (int g = 0; g < 4; ++g) acc[mf][g] = zero;

    // running per-gate B pointers: B[k][n] = W[g*HID + n][k] -> contiguous K runs
    const __bf16* bpH[4];
    const __bf16* bpX[4];
    #pragma unroll
    for (int g = 0; g < 4; ++g) {
        bpH[g] = Whh + (size_t)(g * HID + ncol) * HID  + 16 * half;
        bpX[g] = Wih + (size_t)(g * HID + ncol) * EPAD + 16 * half;
    }

    // prologue: stage chunk 0
    cp_b64_to_lds(&tileA[0][ldOff], srcH);

    // main phase: 31 chunks with next-chunk source always hin (no selects)
    #pragma unroll 1
    for (int kc = 0; kc < 31; ++kc) {
        const int cur = kc & 1;
        wait_async_lds();
        __syncthreads();
        cp_b64_to_lds(&tileA[cur ^ 1][ldOff], srcH + (kc + 1) * 32);
        __builtin_prefetch(bpH[0] + 32, 0, 1);
        mma_chunk(tileA[cur], bpH, acc, l16, half);
        #pragma unroll
        for (int g = 0; g < 4; ++g) bpH[g] += 32;
    }
    // kc = 31: last Whh chunk; stage first x chunk
    wait_async_lds();
    __syncthreads();
    cp_b64_to_lds(&tileA[0][ldOff], srcX);
    mma_chunk(tileA[1], bpH, acc, l16, half);
    // kc = 32: first Wih chunk; stage second x chunk
    wait_async_lds();
    __syncthreads();
    cp_b64_to_lds(&tileA[1][ldOff], srcX + 32);
    mma_chunk(tileA[0], bpX, acc, l16, half);
    // kc = 33: second Wih chunk
    #pragma unroll
    for (int g = 0; g < 4; ++g) bpX[g] += 32;
    wait_async_lds();
    __syncthreads();
    mma_chunk(tileA[1], bpX, acc, l16, half);

    float bv[4];
    #pragma unroll
    for (int g = 0; g < 4; ++g) bv[g] = bias[g * HID + ncol];

    #pragma unroll
    for (int mf = 0; mf < 2; ++mf) {
        #pragma unroll
        for (int v = 0; v < 8; ++v) {
            int brow = mBase + mf * 16 + 8 * half + v;   // C/D layout: M = v + 8*half
            size_t idx = (size_t)brow * HID + ncol;
            float iv = sigmoidf_(acc[mf][0][v] + bv[0]);
            float fv = sigmoidf_(acc[mf][1][v] + bv[1]);
            float gv = tanhf_  (acc[mf][2][v] + bv[2]);
            float ov = sigmoidf_(acc[mf][3][v] + bv[3]);
            float cn = fv * cf32[idx] + iv * gv;
            float hn = ov * tanhf_(cn);
            cf32[idx] = cn;
            hf32[idx] = hn;
            hout[idx] = (__bf16)hn;
        }
    }
}

// ---------------- output projection: scores[t] = h @ W_out.T + b_out ----------------
// grid = BATCH/16, block = 192 (6 waves; wave = V tile of 16)
__global__ __launch_bounds__(192) void k_scores(
    const uint16_t* __restrict__ h_u, const uint16_t* __restrict__ Wout_u,
    const float* __restrict__ bout, float* __restrict__ out)
{
    const __bf16* h    = (const __bf16*)h_u;
    const __bf16* Wout = (const __bf16*)Wout_u;
    const int wave = threadIdx.x >> 5;
    const int lane = threadIdx.x & 31;
    const int l16  = lane & 15;
    const int half = lane >> 4;
    const int mBase = blockIdx.x * 16;
    const int ncol  = wave * 16 + l16;

    const __bf16* ap = h    + (size_t)(mBase + l16) * HID + 16 * half;
    const __bf16* bp = Wout + (size_t)ncol * HID + 16 * half;

    v8f acc = {0.f,0.f,0.f,0.f,0.f,0.f,0.f,0.f};
    #pragma unroll 4
    for (int k0 = 0; k0 < HID; k0 += 32) {
        v8bf lo = *(gv8bf*)(ap);
        v8bf hi = *(gv8bf*)(ap + 8);
        v16bf a = cat8(lo, hi);
        v16bf b = *(gv16bf*)(bp);
        acc = __builtin_amdgcn_wmma_f32_16x16x32_bf16(false, a, false, b,
                                                      (short)0, acc, false, false);
        ap += 32; bp += 32;
    }
    float bb = bout[ncol];
    #pragma unroll
    for (int v = 0; v < 8; ++v) {
        int brow = mBase + 8 * half + v;
        out[(size_t)brow * VOC + ncol] = acc[v] + bb;
    }
}

// ---------------- host ----------------
extern "C" void kernel_launch(void* const* d_in, const int* in_sizes, int n_in,
                              void* d_out, int out_size, void* d_ws, size_t ws_size,
                              hipStream_t stream) {
    (void)in_sizes; (void)n_in; (void)out_size; (void)ws_size;
    const int*   ids  = (const int*)  d_in[0];
    const float* h0   = (const float*)d_in[1];
    const float* c0   = (const float*)d_in[2];
    const float* emb  = (const float*)d_in[3];
    const float* Wih  = (const float*)d_in[4];
    const float* Whh  = (const float*)d_in[5];
    const float* bih  = (const float*)d_in[6];
    const float* bhh  = (const float*)d_in[7];
    const float* Wout = (const float*)d_in[8];
    const float* bout = (const float*)d_in[9];
    float* out = (float*)d_out;

    uint8_t* w = (uint8_t*)d_ws;
    size_t off = 0;
    auto take = [&](size_t bytes) {
        void* p = w + off; off += (bytes + 255) & ~(size_t)255; return p;
    };
    uint16_t* Whh_b  = (uint16_t*)take((size_t)G4 * HID * 2);
    uint16_t* Wih_b  = (uint16_t*)take((size_t)G4 * EPAD * 2);
    uint16_t* Wout_b = (uint16_t*)take((size_t)VOC * HID * 2);
    float*    biasS  = (float*)   take((size_t)G4 * 4);
    uint16_t* xpad   = (uint16_t*)take((size_t)T_STEPS * BATCH * EPAD * 2);
    uint16_t* hb0    = (uint16_t*)take((size_t)BATCH * HID * 2);
    uint16_t* hb1    = (uint16_t*)take((size_t)BATCH * HID * 2);
    float*    hf32   = (float*)   take((size_t)BATCH * HID * 4);
    float*    cf32   = (float*)   take((size_t)BATCH * HID * 4);

    auto g1 = [](int n) { return dim3((unsigned)((n + 255) / 256)); };
    k_cvt_bf16<<<g1(G4 * HID), 256, 0, stream>>>(Whh, Whh_b, G4 * HID);
    k_cvt_bf16<<<g1(VOC * HID), 256, 0, stream>>>(Wout, Wout_b, VOC * HID);
    k_pad_wih<<<g1(G4 * EPAD), 256, 0, stream>>>(Wih, Wih_b);
    k_bias<<<g1(G4), 256, 0, stream>>>(bih, bhh, biasS);
    k_embed<<<g1(T_STEPS * BATCH * EPAD), 256, 0, stream>>>(ids, emb, xpad);
    k_init<<<g1(BATCH * HID), 256, 0, stream>>>(h0, c0, hb0, cf32);

    uint16_t* hbuf[2] = {hb0, hb1};
    for (int t = 0; t < T_STEPS; ++t) {
        const uint16_t* hin = hbuf[t & 1];
        uint16_t* hout = hbuf[(t + 1) & 1];
        dim3 grid(BATCH / 32, HID / 128);
        k_lstm_step<<<grid, 256, 0, stream>>>(Whh_b, Wih_b, biasS,
            xpad + (size_t)t * BATCH * EPAD, hin, hout, hf32, cf32);
        k_scores<<<dim3(BATCH / 16), 192, 0, stream>>>(hout, Wout_b, bout,
            out + (size_t)t * BATCH * VOC);
    }
    k_final<<<g1(BATCH * HID), 256, 0, stream>>>(hf32, cf32,
        out + (size_t)T_STEPS * BATCH * VOC);
}
